// MambaLayer2PE_67001489817882
// MI455X (gfx1250) — matrix operand order repaired and checked by reading, not compiled, not run on hardware
//
#include <hip/hip_runtime.h>
#include <hip/hip_bf16.h>
#include <math.h>

// ---------------- model constants ----------------
#define DIMC      320
#define D_STATE   32
#define D_CONV    4
#define D_INNER   640
#define NHEADS    8
#define HEADDIM   80
#define CONV_DIM  704          // D_INNER + 2*D_STATE
#define D_IN_PROJ 1352         // 2*D_INNER + 2*D_STATE + NHEADS
#define N1_PAD    1408         // D_IN_PROJ padded to /64 for 64x64 block tiles
#define BATCH     2
#define LSEQ      4096         // 64*64
#define MROWS     (BATCH*LSEQ) // 8192
#define LN_EPS    1e-5f
#define RMS_EPS   1e-5f

typedef __attribute__((ext_vector_type(16))) _Float16 v16h;
typedef __attribute__((ext_vector_type(8)))  _Float16 v8h;
typedef __attribute__((ext_vector_type(8)))  float    v8f;

// ---------------- kernel 0: weights -> f16 (in_proj zero-padded to 1408 rows) ----------------
__global__ void prep_weights(const float* __restrict__ w1, const float* __restrict__ w2,
                             _Float16* __restrict__ w1h, _Float16* __restrict__ w2h) {
    const int N1 = N1_PAD * DIMC;
    const int N2 = DIMC * D_INNER;
    int i = blockIdx.x * blockDim.x + threadIdx.x;
    if (i < N1) {
        int n = i / DIMC;
        w1h[i] = (n < D_IN_PROJ) ? (_Float16)w1[i] : (_Float16)0.0f;
    } else if (i < N1 + N2) {
        int j = i - N1;
        w2h[j] = (_Float16)w2[j];
    }
}

// ---------------- kernel 1: pos-embed + LayerNorm -> u16 (f16, [MROWS,320]) ----------------
__global__ void pe_ln_kernel(const float* __restrict__ x, const float* __restrict__ pe_gauss,
                             const float* __restrict__ ln_w, const float* __restrict__ ln_b,
                             _Float16* __restrict__ u16) {
    __shared__ float ssum[DIMC];
    __shared__ float ssq[DIMC];
    __shared__ float s_mu, s_var;

    int row = blockIdx.x;                  // b*4096 + l
    int b   = row >> 12;
    int l   = row & 4095;
    int hy  = l >> 6;
    int wx  = l & 63;
    float cx = 2.0f * ((wx + 0.5f) / 64.0f) - 1.0f;
    float cy = 2.0f * ((hy + 0.5f) / 64.0f) - 1.0f;

    int c = threadIdx.x;                   // 0..319
    int j = (c < 160) ? c : (c - 160);
    float g0 = pe_gauss[j];                // pe_gauss[0][j]
    float g1 = pe_gauss[160 + j];          // pe_gauss[1][j]
    float proj = 6.28318530717958647692f * (cx * g0 + cy * g1);
    float pe = (c < 160) ? __sinf(proj) : __cosf(proj);

    float v = x[(size_t)b * DIMC * LSEQ + (size_t)c * LSEQ + l] + pe;

    ssum[c] = v;
    ssq[c]  = v * v;
    __syncthreads();
    if (c < 160) { ssum[c] += ssum[c+160]; ssq[c] += ssq[c+160]; } __syncthreads();
    if (c <  80) { ssum[c] += ssum[c+ 80]; ssq[c] += ssq[c+ 80]; } __syncthreads();
    if (c <  40) { ssum[c] += ssum[c+ 40]; ssq[c] += ssq[c+ 40]; } __syncthreads();
    if (c <  20) { ssum[c] += ssum[c+ 20]; ssq[c] += ssq[c+ 20]; } __syncthreads();
    if (c <  10) { ssum[c] += ssum[c+ 10]; ssq[c] += ssq[c+ 10]; } __syncthreads();
    if (c <   5) { ssum[c] += ssum[c+  5]; ssq[c] += ssq[c+  5]; } __syncthreads();
    if (c == 0) {
        float a = 0.f, q = 0.f;
        for (int i = 0; i < 5; ++i) { a += ssum[i]; q += ssq[i]; }
        float mu = a / (float)DIMC;
        s_mu  = mu;
        s_var = q / (float)DIMC - mu * mu;
    }
    __syncthreads();
    float un = (v - s_mu) * rsqrtf(s_var + LN_EPS) * ln_w[c] + ln_b[c];
    u16[(size_t)row * DIMC + c] = (_Float16)un;
}

// ---------------- WMMA fragment helpers ----------------
// A-matrix 16x32 f16 fragment per ISA layout: lane m = lane&15,
// elems 0..7 = K[kb + 8*half + j], elems 8..15 = K[kb + 16 + 8*half + j]
__device__ __forceinline__ v16h load_a_frag(const _Float16* arow, int kb, int half) {
    v8h lo = *(const v8h*)(arow + kb + half * 8);
    v8h hi = *(const v8h*)(arow + kb + 16 + half * 8);
    return __builtin_shufflevector(lo, hi, 0,1,2,3,4,5,6,7,8,9,10,11,12,13,14,15);
}

// ---------------- register-blocked WMMA GEMM core: 32x32 output per wave ----------------
// D[m0..m0+31, n0..n0+31] = A[m0..,:K] x W[n0..,:K]^T  (f16 in, f32 acc)
template <int K>
__device__ __forceinline__ void gemm_tile_32x32(const _Float16* __restrict__ A,
                                                const _Float16* __restrict__ W,
                                                int m0, int n0, int lane,
                                                v8f& acc00, v8f& acc01,
                                                v8f& acc10, v8f& acc11) {
    int half = lane >> 4;
    int lm   = lane & 15;
    const _Float16* arow0 = A + (size_t)(m0 + lm) * K;
    const _Float16* arow1 = A + (size_t)(m0 + 16 + lm) * K;
    const _Float16* wrow0 = W + (size_t)(n0 + lm) * K;
    const _Float16* wrow1 = W + (size_t)(n0 + 16 + lm) * K;
    __builtin_prefetch(arow0, 0, 1);
    __builtin_prefetch(arow1, 0, 1);
    __builtin_prefetch(wrow0, 0, 1);
    __builtin_prefetch(wrow1, 0, 1);
    #pragma unroll
    for (int kb = 0; kb < K; kb += 32) {
        v16h a0 = load_a_frag(arow0, kb, half);
        v16h a1 = load_a_frag(arow1, kb, half);
        v16h b0 = load_a_frag(wrow0, kb, half);
        v16h b1 = load_a_frag(wrow1, kb, half);
        acc00 = __builtin_amdgcn_wmma_f32_16x16x32_f16(false, a0, false, b0, (short)0, acc00, false, false);
        acc01 = __builtin_amdgcn_wmma_f32_16x16x32_f16(false, a0, false, b1, (short)0, acc01, false, false);
        acc10 = __builtin_amdgcn_wmma_f32_16x16x32_f16(false, a1, false, b0, (short)0, acc10, false, false);
        acc11 = __builtin_amdgcn_wmma_f32_16x16x32_f16(false, a1, false, b1, (short)0, acc11, false, false);
    }
}

// ---------------- kernel 2: GEMM1  zxbcdt[M,1408] = u16[M,320] @ W1^T ----------------
// block = 128 threads = 4 waves in 2x2; block tile 64x64; wave tile 32x32
__global__ void gemm1_kernel(const _Float16* __restrict__ A, const _Float16* __restrict__ W,
                             float* __restrict__ C) {
    const int K = DIMC;                    // 320
    int wave = threadIdx.x >> 5;
    int lane = threadIdx.x & 31;
    int m0 = blockIdx.x * 64 + (wave >> 1) * 32;
    int n0 = blockIdx.y * 64 + (wave & 1) * 32;

    v8f acc00 = {}, acc01 = {}, acc10 = {}, acc11 = {};
    gemm_tile_32x32<K>(A, W, m0, n0, lane, acc00, acc01, acc10, acc11);

    int half = lane >> 4;
    int lm   = lane & 15;
    float* out = C + (size_t)(m0 + half * 8) * N1_PAD + n0 + lm;
    #pragma unroll
    for (int r = 0; r < 8; ++r) {
        out[(size_t)r * N1_PAD]      = acc00[r];
        out[(size_t)r * N1_PAD + 16] = acc01[r];
        out[(size_t)(r + 16) * N1_PAD]      = acc10[r];
        out[(size_t)(r + 16) * N1_PAD + 16] = acc11[r];
    }
}

// ---------------- kernel 3: depthwise conv(4) + SiLU -> xBC [B,L,704] ----------------
__global__ void conv_silu_kernel(const float* __restrict__ zxbcdt,
                                 const float* __restrict__ conv_w,
                                 const float* __restrict__ conv_b,
                                 float* __restrict__ xBC) {
    int idx = blockIdx.x * blockDim.x + threadIdx.x;
    int total = BATCH * LSEQ * CONV_DIM;
    if (idx >= total) return;
    int c  = idx % CONV_DIM;
    int bt = idx / CONV_DIM;               // b*4096 + t
    int t  = bt & 4095;
    float acc = conv_b[c];
    #pragma unroll
    for (int k = 0; k < D_CONV; ++k) {
        int tt = t - (D_CONV - 1) + k;
        if (tt >= 0) {
            int bt2 = bt - (t - tt);
            acc += zxbcdt[(size_t)bt2 * N1_PAD + D_INNER + c] * conv_w[c * D_CONV + k];
        }
    }
    acc = acc / (1.0f + __expf(-acc));     // SiLU
    xBC[(size_t)bt * CONV_DIM + c] = acc;
}

// ---------------- kernel 4: sequential SSM scan, one block per (b,h) ----------------
__global__ void scan_kernel(const float* __restrict__ zxbcdt,
                            const float* __restrict__ xBC,
                            const float* __restrict__ dt_bias,
                            const float* __restrict__ A_log,
                            const float* __restrict__ Dp,
                            float* __restrict__ ybuf) {
    int bh = blockIdx.x;
    int b  = bh >> 3;
    int h  = bh & 7;
    int p  = threadIdx.x;                  // 0..95, active 0..79

    __shared__ float Bs[D_STATE];
    __shared__ float Cs[D_STATE];

    float st[D_STATE];
    #pragma unroll
    for (int n = 0; n < D_STATE; ++n) st[n] = 0.0f;

    float Ah    = -__expf(A_log[h]);
    float dbias = dt_bias[h];
    float dph   = Dp[h];

    for (int t = 0; t < LSEQ; ++t) {
        size_t bt = (size_t)b * LSEQ + t;
        const float* row = xBC + bt * CONV_DIM;
        if (threadIdx.x < 32)       Bs[threadIdx.x]      = row[D_INNER + threadIdx.x];
        else if (threadIdx.x < 64)  Cs[threadIdx.x - 32] = row[D_INNER + threadIdx.x];
        __syncthreads();

        float draw = zxbcdt[bt * N1_PAD + (D_INNER + CONV_DIM) + h] + dbias;
        float dtv  = (draw > 20.0f) ? draw : log1pf(__expf(draw));
        float dA   = __expf(dtv * Ah);

        if (p < HEADDIM) {
            float xv  = row[h * HEADDIM + p];
            float dtx = dtv * xv;
            float y = 0.0f;
            #pragma unroll
            for (int n = 0; n < D_STATE; ++n) {
                st[n] = st[n] * dA + dtx * Bs[n];
                y += st[n] * Cs[n];
            }
            ybuf[bt * D_INNER + h * HEADDIM + p] = y + dph * xv;
        }
        __syncthreads();
    }
}

// ---------------- kernel 5: gate with SiLU(z) + RMSNorm -> y16 (f16 [M,640]) ----------------
__global__ void gate_rms_kernel(const float* __restrict__ zxbcdt,
                                float* __restrict__ ybuf,
                                const float* __restrict__ rms_w,
                                _Float16* __restrict__ y16) {
    __shared__ float red[256];
    __shared__ float s_scale;
    int m = blockIdx.x;
    int tid = threadIdx.x;

    float local = 0.0f;
    for (int c = tid; c < D_INNER; c += 256) {
        float z = zxbcdt[(size_t)m * N1_PAD + c];
        float g = ybuf[(size_t)m * D_INNER + c] * (z / (1.0f + __expf(-z)));
        ybuf[(size_t)m * D_INNER + c] = g;
        local += g * g;
    }
    red[tid] = local;
    __syncthreads();
    for (int off = 128; off > 0; off >>= 1) {
        if (tid < off) red[tid] += red[tid + off];
        __syncthreads();
    }
    if (tid == 0) s_scale = rsqrtf(red[0] / (float)D_INNER + RMS_EPS);
    __syncthreads();
    float s = s_scale;
    for (int c = tid; c < D_INNER; c += 256) {
        y16[(size_t)m * D_INNER + c] = (_Float16)(ybuf[(size_t)m * D_INNER + c] * s * rms_w[c]);
    }
}

// ---------------- kernel 6: GEMM2  out[b,n,l] = y16[M,640] @ W2^T (transposed store) ----------------
__global__ void gemm2_kernel(const _Float16* __restrict__ A, const _Float16* __restrict__ W,
                             float* __restrict__ out) {
    const int K = D_INNER;                 // 640
    int wave = threadIdx.x >> 5;
    int lane = threadIdx.x & 31;
    int m0 = blockIdx.x * 64 + (wave >> 1) * 32;
    int n0 = blockIdx.y * 64 + (wave & 1) * 32;

    v8f acc00 = {}, acc01 = {}, acc10 = {}, acc11 = {};
    gemm_tile_32x32<K>(A, W, m0, n0, lane, acc00, acc01, acc10, acc11);

    int half = lane >> 4;
    int lm   = lane & 15;
    #pragma unroll
    for (int r = 0; r < 8; ++r) {
        #pragma unroll
        for (int sm = 0; sm < 2; ++sm) {
            int m  = m0 + sm * 16 + r + 8 * half;
            int bb = m >> 12;
            int t  = m & 4095;
            size_t base = (size_t)bb * DIMC * LSEQ + (size_t)t;
            out[base + (size_t)(n0 + lm) * LSEQ]      = (sm == 0) ? acc00[r] : acc10[r];
            out[base + (size_t)(n0 + 16 + lm) * LSEQ] = (sm == 0) ? acc01[r] : acc11[r];
        }
    }
}

// ---------------- launch ----------------
extern "C" void kernel_launch(void* const* d_in, const int* in_sizes, int n_in,
                              void* d_out, int out_size, void* d_ws, size_t ws_size,
                              hipStream_t stream) {
    const float* x          = (const float*)d_in[0];
    const float* pe_gauss   = (const float*)d_in[1];
    const float* ln_w       = (const float*)d_in[2];
    const float* ln_b       = (const float*)d_in[3];
    const float* in_proj_w  = (const float*)d_in[4];
    const float* conv_w     = (const float*)d_in[5];
    const float* conv_b     = (const float*)d_in[6];
    const float* dt_bias    = (const float*)d_in[7];
    const float* A_log      = (const float*)d_in[8];
    const float* Dp         = (const float*)d_in[9];
    const float* rms_w      = (const float*)d_in[10];
    const float* out_proj_w = (const float*)d_in[11];
    float* out = (float*)d_out;

    char* ws = (char*)d_ws;
    _Float16* u16   = (_Float16*)ws;  ws += (size_t)MROWS * DIMC * 2;        // 5.24 MB
    _Float16* w1h   = (_Float16*)ws;  ws += (size_t)N1_PAD * DIMC * 2;       // 0.90 MB
    _Float16* w2h   = (_Float16*)ws;  ws += (size_t)DIMC * D_INNER * 2;      // 0.41 MB
    float*    zxb   = (float*)ws;     ws += (size_t)MROWS * N1_PAD * 4;      // 46.1 MB
    float*    xBC   = (float*)ws;     ws += (size_t)MROWS * CONV_DIM * 4;    // 23.1 MB
    float*    ybuf  = (float*)ws;     ws += (size_t)MROWS * D_INNER * 4;     // 21.0 MB
    _Float16* y16   = (_Float16*)ws;  ws += (size_t)MROWS * D_INNER * 2;     // 10.5 MB

    // 0) weights -> f16 (zero-padded in_proj)
    {
        int total = N1_PAD * DIMC + DIMC * D_INNER;
        prep_weights<<<(total + 255) / 256, 256, 0, stream>>>(in_proj_w, out_proj_w, w1h, w2h);
    }
    // 1) pos-embed + LayerNorm
    pe_ln_kernel<<<MROWS, DIMC, 0, stream>>>(x, pe_gauss, ln_w, ln_b, u16);
    // 2) in_proj GEMM (WMMA, 32x32/wave): [8192,320] x [1408,320]^T
    gemm1_kernel<<<dim3(MROWS / 64, N1_PAD / 64), 128, 0, stream>>>(u16, w1h, zxb);
    // 3) depthwise conv + SiLU
    {
        int total = BATCH * LSEQ * CONV_DIM;
        conv_silu_kernel<<<(total + 255) / 256, 256, 0, stream>>>(zxb, conv_w, conv_b, xBC);
    }
    // 4) sequential SSM scan per (b,h)
    scan_kernel<<<BATCH * NHEADS, 96, 0, stream>>>(zxb, xBC, dt_bias, A_log, Dp, ybuf);
    // 5) gate + RMSNorm -> f16
    gate_rms_kernel<<<MROWS, 256, 0, stream>>>(zxb, ybuf, rms_w, y16);
    // 6) out_proj GEMM (WMMA, 32x32/wave) with transposed store into d_out [B,C,H,W]
    gemm2_kernel<<<dim3(MROWS / 64, DIMC / 64), 128, 0, stream>>>(y16, w2h, out);
}
